// MoELayerRouterModel_19825569038532
// MI455X (gfx1250) — matrix operand order repaired and checked by reading, not compiled, not run on hardware
//
#include <hip/hip_runtime.h>
#include <hip/hip_bf16.h>

typedef __attribute__((ext_vector_type(16))) __bf16 v16bf;
typedef __attribute__((ext_vector_type(8)))  float  v8f;
typedef __attribute__((ext_vector_type(4)))  unsigned int u32x4;
typedef __attribute__((ext_vector_type(8)))  int i32x8;
typedef __attribute__((ext_vector_type(4)))  int i32x4;

#define N_ROWS 16384
#define DIM_D  4096
#define DIM_H  1024
#define DIM_E  64
#define TOPK   8
#define CLAMP_MAX 4.60517018598809136804f  // log(100)
#define NORM_EPS  1e-12f
#define LDPAD 40     // gemm1 LDS row: 32 bf16 + 8 pad (80B, 16B-aligned)
#define BPAD  264    // gemm2 LDS row: 256 bf16 + 8 pad (528B) -- matches TDM pad

// ---- fragment loaders -------------------------------------------------------
// A-matrix 16x32 bf16 (ISA 7.12.2): lane L holds row M=L%16; K chunks
// [half*8 .. half*8+7] and [half*8+16 .. half*8+23]  (half = L/16)
__device__ __forceinline__ v16bf load_afrag(const __bf16* row_base, int half) {
  union { v16bf v; uint4 q[2]; } u;
  const __bf16* p = row_base + half * 8;
  u.q[0] = *(const uint4*)(p);
  u.q[1] = *(const uint4*)(p + 16);
  return u.v;
}
// B-matrix 32x16 bf16: lane L holds col N=L%16; K chunks
// [half*8 .. half*8+7] and [half*8+16 .. half*8+23]  (half = L/16)
__device__ __forceinline__ v16bf load_bfrag(const __bf16* col_base, int half) {
  union { v16bf v; uint4 q[2]; } u;
  const __bf16* p = col_base + half * 8;
  u.q[0] = *(const uint4*)(p);
  u.q[1] = *(const uint4*)(p + 16);
  return u.v;
}

__device__ __forceinline__ v8f wmma_bf16(v16bf a, v16bf b, v8f c) {
  return __builtin_amdgcn_wmma_f32_16x16x32_bf16(false, a, false, b, (short)0, c, false, false);
}

// pack 4 floats -> 4 bf16 -> one 8-byte LDS store
__device__ __forceinline__ void store4bf(__bf16* dst, float a, float b, float c, float d) {
  union { __bf16 h[4]; unsigned long long u; } p;
  p.h[0] = (__bf16)a; p.h[1] = (__bf16)b; p.h[2] = (__bf16)c; p.h[3] = (__bf16)d;
  *(unsigned long long*)dst = p.u;
}

// ---- K0: zero accumulators --------------------------------------------------
__global__ void init_kernel(float* __restrict__ rowsq, float* __restrict__ impload) {
  int i = blockIdx.x * 256 + threadIdx.x;
  if (i < N_ROWS) rowsq[i] = 0.0f;
  if (i < 2 * DIM_E) impload[i] = 0.0f;
}

// ---- K1: simT[e][h] = bf16( sim[:,e] / max(||sim[:,e]||, eps) ) -------------
__global__ __launch_bounds__(256) void simnorm_kernel(const float* __restrict__ sim,
                                                      __bf16* __restrict__ simT) {
  int e = blockIdx.x;
  int t = threadIdx.x;
  float s = 0.0f;
  for (int i = t; i < DIM_H; i += 256) {
    float v = sim[(size_t)i * DIM_E + e];
    s += v * v;
  }
  __shared__ float red[256];
  red[t] = s;
  __syncthreads();
  for (int off = 128; off > 0; off >>= 1) {
    if (t < off) red[t] += red[t + off];
    __syncthreads();
  }
  __shared__ float invn;
  if (t == 0) invn = 1.0f / fmaxf(sqrtf(red[0]), NORM_EPS);
  __syncthreads();
  float iv = invn;
  for (int i = t; i < DIM_H; i += 256)
    simT[(size_t)e * DIM_H + i] = (__bf16)(sim[(size_t)i * DIM_E + e] * iv);
}

// ---- K2: GEMM1  h = x @ W + b  (bf16 out) + rowsq accumulation -------------
// block: 128x128 tile, 256 thr = 8 waves (2x4); wave: 64x32 = 4x2 frags
// double-buffered LDS + register prefetch; packed b64 LDS staging stores
__global__ __launch_bounds__(256) void gemm1_kernel(const float* __restrict__ x,
                                                    const float* __restrict__ W,
                                                    const float* __restrict__ bias,
                                                    __bf16* __restrict__ hout,
                                                    float* __restrict__ rowsq) {
  __shared__ __bf16 As[2][128 * LDPAD];
  __shared__ __bf16 Bs[2][128 * LDPAD];

  const int bm = blockIdx.x;          // 0..127  (rows / 128)
  const int bn = blockIdx.y;          // 0..7    (cols / 128)
  const int t = threadIdx.x;
  const int wave = t >> 5, lane = t & 31;
  const int wm = wave >> 2, wn = wave & 3;    // 2 x 4 wave grid
  const int half = lane >> 4, m = lane & 15;

  // A staging: thread handles 4 float4s (row = f/8, col4 = f%8)
  const int a_row[4] = { (t + 0) >> 3, (t + 256) >> 3, (t + 512) >> 3, (t + 768) >> 3 };
  const int a_c4 = (t & 7) * 4;
  // B staging: thread owns column n, one k-quad per 8-k group
  const int b_n = t & 127;
  const int b_kh = (t >> 7) * 4;      // 0 or 4

  v8f acc[4][2] = {};
  float4 ax[4];
  float bw[4][4];

  auto prefetch = [&](int ks) {
#pragma unroll
    for (int c = 0; c < 4; ++c)
      ax[c] = *(const float4*)(x + (size_t)(bm * 128 + a_row[c]) * DIM_D + ks + a_c4);
#pragma unroll
    for (int g = 0; g < 4; ++g) {
      int k = g * 8 + b_kh;
#pragma unroll
      for (int i = 0; i < 4; ++i)
        bw[g][i] = W[(size_t)(ks + k + i) * DIM_H + bn * 128 + b_n];
    }
  };
  auto stage = [&](int buf) {
#pragma unroll
    for (int c = 0; c < 4; ++c)
      store4bf(&As[buf][a_row[c] * LDPAD + a_c4], ax[c].x, ax[c].y, ax[c].z, ax[c].w);
#pragma unroll
    for (int g = 0; g < 4; ++g)
      store4bf(&Bs[buf][b_n * LDPAD + g * 8 + b_kh], bw[g][0], bw[g][1], bw[g][2], bw[g][3]);
  };

  prefetch(0);
  stage(0);

  const int nsteps = DIM_D / 32;
  for (int s = 0; s < nsteps; ++s) {
    const int buf = s & 1;
    __syncthreads();
    if (s + 1 < nsteps) prefetch((s + 1) * 32);

    v16bf a[4], b[2];
#pragma unroll
    for (int i = 0; i < 4; ++i)
      a[i] = load_afrag(&As[buf][(wm * 64 + i * 16 + m) * LDPAD], half);
#pragma unroll
    for (int j = 0; j < 2; ++j)
      b[j] = load_bfrag(&Bs[buf][(wn * 32 + j * 16 + m) * LDPAD], half);
#pragma unroll
    for (int i = 0; i < 4; ++i)
#pragma unroll
      for (int j = 0; j < 2; ++j)
        acc[i][j] = wmma_bf16(a[i], b[j], acc[i][j]);

    if (s + 1 < nsteps) stage(buf ^ 1);
  }

  // epilogue: +bias, store bf16 h, accumulate row sum-of-squares
  float bv[2];
#pragma unroll
  for (int j = 0; j < 2; ++j)
    bv[j] = bias[bn * 128 + wn * 32 + j * 16 + m];

#pragma unroll
  for (int i = 0; i < 4; ++i) {
#pragma unroll
    for (int r = 0; r < 8; ++r) {
      int row = bm * 128 + wm * 64 + i * 16 + r + half * 8;
      float ss = 0.0f;
#pragma unroll
      for (int j = 0; j < 2; ++j) {
        int col = bn * 128 + wn * 32 + j * 16 + m;
        float hv = acc[i][j][r] + bv[j];
        hout[(size_t)row * DIM_H + col] = (__bf16)hv;
        ss += hv * hv;
      }
      ss += __shfl_xor(ss, 1, 32);
      ss += __shfl_xor(ss, 2, 32);
      ss += __shfl_xor(ss, 4, 32);
      ss += __shfl_xor(ss, 8, 32);
      if (m == 0) atomicAdd(&rowsq[row], ss);
    }
  }
}

// ---- TDM: stage a 64-row x 256-col bf16 tile of simT into LDS --------------
// D# per CDNA5 ISA: 2D tensor, data_size=2B, LDS padding 4 dwords / 128 dwords
__device__ __forceinline__ void tdm_load_simT_tile(const __bf16* simT, int ks0,
                                                   unsigned int lds_byte_addr) {
  unsigned long long ga = (unsigned long long)(uintptr_t)(simT + ks0);
  u32x4 g0;
  g0.x = 1u;                                               // count=1, user mode
  g0.y = lds_byte_addr;                                    // lds_addr
  g0.z = (unsigned int)ga;                                 // global_addr[31:0]
  g0.w = (unsigned int)((ga >> 32) & 0x1FFFFFFu) | (2u << 30); // addr[56:32] | type=2
  i32x8 g1;
  g1[0] = (1 << 16)      // data_size = 2 bytes
        | (1 << 20)      // pad_enable
        | (6 << 22)      // pad_interval: 128 dwords (one 512B row)
        | (3 << 25);     // pad_amount: 4 dwords (16B) -> 528B row stride
  g1[1] = (int)((DIM_H & 0xFFFFu) << 16);                  // tensor_dim0 lo16
  g1[2] = (int)((DIM_H >> 16) | ((DIM_E & 0xFFFF) << 16)); // dim0 hi16 | dim1 lo16
  g1[3] = (int)((DIM_E >> 16) | (256 << 16));              // dim1 hi16 | tile_dim0=256
  g1[4] = (int)(DIM_E);                                    // tile_dim1=64, tile_dim2=0
  g1[5] = DIM_H;                                           // tensor_dim0_stride lo32
  g1[6] = 0;                                               // stride0 hi16 | stride1 lo16
  g1[7] = 0;
  i32x4 z4 = {0, 0, 0, 0};
  i32x8 z8 = {0, 0, 0, 0, 0, 0, 0, 0};
  __builtin_amdgcn_tensor_load_to_lds(g0, g1, z4, z4, z8, 0);
}

// ---- K3: GEMM2 (h @ simT^T) + scale + top-8 + softmax + reductions ---------
// block: 128 rows (8 waves x 16 rows); wave: 16x64 logits = 4 frags, K=1024
// B staged by TDM in 4 chunks of 64x256; logits scratch reuses the same LDS.
__global__ __launch_bounds__(256) void gemm2_topk_kernel(const __bf16* __restrict__ h,
                                                         const __bf16* __restrict__ simT,
                                                         const float* __restrict__ rowsq,
                                                         const float* __restrict__ temperature,
                                                         float* __restrict__ mask_out,
                                                         float* __restrict__ prob_out,
                                                         float* __restrict__ importance,
                                                         float* __restrict__ load_out) {
  __shared__ __align__(16) char smraw[DIM_E * BPAD * 2];   // 33792B: B tile / logits
  __bf16* Bsm = (__bf16*)smraw;
  float*  lg  = (float*)smraw;                             // [128][64] phase-2 view

  const int t = threadIdx.x;
  const int wave = t >> 5, lane = t & 31;
  const int half = lane >> 4, m = lane & 15;
  const int rowbase = blockIdx.x * 128 + wave * 16;

  v8f acc[4] = {};
  const __bf16* arow = h + (size_t)(rowbase + m) * DIM_H;

  for (int chunk = 0; chunk < 4; ++chunk) {
    const int ks0 = chunk * 256;
    __syncthreads();                         // prior chunk readers done
    if (wave == 0) {
      tdm_load_simT_tile(simT, ks0, (unsigned int)(uintptr_t)Bsm);
      __builtin_amdgcn_s_wait_tensorcnt(0);
    }
    __syncthreads();                         // tile visible to all waves
#pragma unroll
    for (int ks = 0; ks < 256; ks += 32) {
      v16bf a = load_afrag(arow + ks0 + ks, half);
#pragma unroll
      for (int j = 0; j < 4; ++j) {
        v16bf b = load_bfrag(Bsm + (size_t)(j * 16 + m) * BPAD + ks, half);
        acc[j] = wmma_bf16(a, b, acc[j]);
      }
    }
  }
  __syncthreads();                           // done with B tile; reuse LDS as lg

  const float scale = __expf(fminf(temperature[0], CLAMP_MAX));

  // scaled logits -> LDS [row][expert]
#pragma unroll
  for (int j = 0; j < 4; ++j) {
#pragma unroll
    for (int r = 0; r < 8; ++r) {
      int rr = r + half * 8;
      float nrm = sqrtf(rowsq[rowbase + rr]);
      lg[(size_t)(wave * 16 + rr) * DIM_E + j * 16 + m] =
          acc[j][r] * scale / fmaxf(nrm, NORM_EPS);
    }
  }
  __syncthreads();

  float imp0 = 0.0f, imp1 = 0.0f, c0 = 0.0f, c1 = 0.0f;
  for (int rr = 0; rr < 16; ++rr) {
    int row = rowbase + rr;
    float v0 = lg[(size_t)(wave * 16 + rr) * DIM_E + lane];        // experts 0..31
    float v1 = lg[(size_t)(wave * 16 + rr) * DIM_E + lane + 32];   // experts 32..63
    float w0 = v0, w1 = v1;
    float m0 = 0.0f, m1 = 0.0f;
    float rowmax = 0.0f;
#pragma unroll
    for (int kk = 0; kk < TOPK; ++kk) {
      float bvv = (w0 >= w1) ? w0 : w1;    // tie -> lower index
      int bi = (w0 >= w1) ? lane : lane + 32;
#pragma unroll
      for (int off = 16; off > 0; off >>= 1) {
        float ov = __shfl_xor(bvv, off, 32);
        int oi = __shfl_xor(bi, off, 32);
        if (ov > bvv || (ov == bvv && oi < bi)) { bvv = ov; bi = oi; }
      }
      if (kk == 0) rowmax = bvv;
      if (bi == lane)      { m0 = 1.0f; w0 = -__builtin_inff(); }
      if (bi == lane + 32) { m1 = 1.0f; w1 = -__builtin_inff(); }
    }
    float e0 = __expf(v0 - rowmax);
    float e1 = __expf(v1 - rowmax);
    float s = e0 + e1;
#pragma unroll
    for (int off = 16; off > 0; off >>= 1) s += __shfl_xor(s, off, 32);
    float inv = 1.0f / s;
    float p0 = e0 * inv, p1 = e1 * inv;
    prob_out[(size_t)row * DIM_E + lane]      = p0;
    prob_out[(size_t)row * DIM_E + lane + 32] = p1;
    mask_out[(size_t)row * DIM_E + lane]      = m0;
    mask_out[(size_t)row * DIM_E + lane + 32] = m1;
    imp0 += p0; imp1 += p1; c0 += m0; c1 += m1;
  }
  atomicAdd(&importance[lane],      imp0);
  atomicAdd(&importance[lane + 32], imp1);
  atomicAdd(&load_out[lane],        c0);
  atomicAdd(&load_out[lane + 32],   c1);
}

// ---- launcher ---------------------------------------------------------------
extern "C" void kernel_launch(void* const* d_in, const int* in_sizes, int n_in,
                              void* d_out, int out_size, void* d_ws, size_t ws_size,
                              hipStream_t stream) {
  (void)in_sizes; (void)n_in; (void)out_size; (void)ws_size;
  const float* x    = (const float*)d_in[0];
  const float* W    = (const float*)d_in[1];
  const float* bias = (const float*)d_in[2];
  const float* sim  = (const float*)d_in[3];
  const float* temp = (const float*)d_in[4];

  float* out        = (float*)d_out;
  float* mask_out   = out;                                   // [N,64]
  float* prob_out   = out + (size_t)N_ROWS * DIM_E;          // [N,64]
  float* importance = out + (size_t)2 * N_ROWS * DIM_E;      // [64] (+64 load)

  char* ws = (char*)d_ws;
  __bf16* hbuf   = (__bf16*)ws;                                         // 32 MB
  __bf16* simT   = (__bf16*)(ws + (size_t)N_ROWS * DIM_H * 2);          // 128 KB
  float*  rowsq  = (float*)(ws + (size_t)N_ROWS * DIM_H * 2 + (size_t)DIM_E * DIM_H * 2);

  init_kernel<<<64, 256, 0, stream>>>(rowsq, importance);
  simnorm_kernel<<<DIM_E, 256, 0, stream>>>(sim, simT);
  dim3 g1(N_ROWS / 128, DIM_H / 128);
  gemm1_kernel<<<g1, 256, 0, stream>>>(x, W, bias, hbuf, rowsq);
  gemm2_topk_kernel<<<N_ROWS / 128, 256, 0, stream>>>(hbuf, simT, rowsq, temp,
                                                      mask_out, prob_out,
                                                      importance, importance + DIM_E);
}
